// HierarchicalNetwork_32564442038350
// MI455X (gfx1250) — matrix operand order repaired and checked by reading, not compiled, not run on hardware
//
#include <hip/hip_runtime.h>
#include <hip/hip_bf16.h>
#include <cstdint>

// ---------------------------------------------------------------------------
// MI455X (gfx1250, CDNA5) Bayesian transformer.
// - All GEMM-shaped math through V_WMMA_F32_16X16X32_BF16 (wave32 WMMA).
// - GEMM K-tiles double-buffered in LDS via GLOBAL_LOAD_ASYNC_TO_LDS_B128
//   (ASYNCcnt) when the toolchain exposes the builtins; sync fallback else.
// ---------------------------------------------------------------------------

#define B_    512
#define L_    128
#define D_    128
#define H_    4
#define DK_   32
#define DFF_  256
#define T_    2048
#define NT_   (B_ * L_)          // 65536 tokens

typedef __bf16 bf16_t;
typedef __attribute__((ext_vector_type(16))) __bf16 v16bf;
typedef __attribute__((ext_vector_type(8)))  float  v8f;

// ------------------------------ async global->LDS copies -------------------
#if __has_builtin(__builtin_amdgcn_global_load_async_to_lds_b128) && \
    __has_builtin(__builtin_amdgcn_s_wait_asynccnt)
#define ASYNC_LDS 1
#else
#define ASYNC_LDS 0
#endif

#if ASYNC_LDS
typedef __attribute__((__vector_size__(4 * sizeof(int)))) int v4i_t;
__device__ __forceinline__ void cp_async_b128(const void* g, const void* l) {
  __builtin_amdgcn_global_load_async_to_lds_b128(
      (v4i_t*)(uintptr_t)g, (v4i_t*)(uintptr_t)l, 0, 0);
}
template<int N> __device__ __forceinline__ void cp_wait() {
  __builtin_amdgcn_s_wait_asynccnt(N);
}
#else
template<int N> __device__ __forceinline__ void cp_wait() {}
#endif

// ------------------------------ RNG (stateless, deterministic) -------------
__device__ __forceinline__ unsigned hash_u32(unsigned x) {
  x ^= x >> 16; x *= 0x7feb352du;
  x ^= x >> 15; x *= 0x846ca68bu;
  x ^= x >> 16; return x;
}
__device__ __forceinline__ float rnorm_hash(unsigned seed, unsigned idx) {
  unsigned h1 = hash_u32(seed * 0x9E3779B9u ^ (idx * 2u + 0x68bc21ebu));
  unsigned h2 = hash_u32(seed ^ (idx * 2u + 1u) * 0x85EBCA6Bu);
  float u1 = ((float)h1 + 1.0f) * 2.3283064e-10f;   // (0,1]
  float u2 = (float)h2 * 2.3283064e-10f;
  return sqrtf(-2.0f * __logf(u1)) * __cosf(6.2831853f * u2);
}

// ------------------------------ sampling / conversion ----------------------
__global__ void k_sample_bf16(const float* __restrict__ mu, const float* __restrict__ ls,
                              bf16_t* __restrict__ out, int n, unsigned seed) {
  int i = blockIdx.x * 256 + threadIdx.x;
  if (i < n) out[i] = (bf16_t)(mu[i] + __expf(ls[i]) * rnorm_hash(seed, (unsigned)i));
}
__global__ void k_sample_f32(const float* __restrict__ mu, const float* __restrict__ ls,
                             float* __restrict__ out, int n, unsigned seed) {
  int i = blockIdx.x * 256 + threadIdx.x;
  if (i < n) out[i] = mu[i] + __expf(ls[i]) * rnorm_hash(seed, (unsigned)i);
}
__global__ void k_cvt_bf16(const float* __restrict__ in, bf16_t* __restrict__ out, int n) {
  int i = blockIdx.x * 256 + threadIdx.x;
  if (i < n) out[i] = (bf16_t)in[i];
}

// ------------------------------ embedding + positional encoding ------------
__global__ void k_embed_pos(const int* __restrict__ ev, const int* __restrict__ mask,
                            const float* __restrict__ emb,
                            float* __restrict__ X, float* __restrict__ INP) {
  int i = blockIdx.x * 256 + threadIdx.x;          // over NT_*D_
  if (i >= NT_ * D_) return;
  int n = i >> 7;                                  // / D_
  int d = i & (D_ - 1);
  float e  = emb[(size_t)ev[n] * D_ + d];
  int  mk  = mask[n];
  float ang = (float)mk * __expf((float)d * (-9.2103404f / (float)D_)); // ln(1e4)
  float pe  = ((mk & 1) == 0) ? __sinf(ang) : __cosf(ang);
  X[i]   = e;
  INP[i] = e + pe;
}

// ------------------------------ WMMA fragment helpers ----------------------
// 16-bit A-matrix 16x32 k order used here (contiguous, not AMD's table order): g=lane>>4, j=0..15:
//   K = g*16 + j
__device__ __forceinline__ int a_k_of(int j, int grp) {
  return grp * 16 + j;
}
// B-matrix 32x16: half j of lane holds (K = grp*16 + j, N = lane&15).
// C/D f32 16x16: VGPR v, lanes 0-15 -> (M=v, N=lane); lanes 16-31 -> M=8+v.

// ------------------------------ GEMM: C = A(f32,M x K) * W^T(bf16,N x K) + b
// Block: 128 threads = 4 waves; 64x32 C tile (waves stacked in M, each wave
// owns a 16x32 strip -> 2 WMMAs per K-step). K-tiles double-buffered in LDS;
// with ASYNC_LDS each wave issues exactly 5 async b128 copies per tile, so
// "s_wait_asynccnt 5" retires the current tile while the next stays in flight.
#define ASTR_G 36     // f32 stride  (144 B: 16-B aligned, bank-spread)
#define BSTR_G 40     // bf16 stride ( 80 B: 16-B aligned, bank-spread)

template<bool RELU>
__global__ void __launch_bounds__(128) k_gemm_bf16(
    const float* __restrict__ A, int lda,
    const bf16_t* __restrict__ W,           // row-major (Nout, K)
    const float* __restrict__ bias,
    float* __restrict__ C, int K, int Nout)
{
  const int tid  = threadIdx.x;
  const int wave = tid >> 5;
  const int lane = tid & 31;
  const int n0 = blockIdx.x * 32;
  const int m0 = blockIdx.y * 64;

  __shared__ float  Asf[2][64 * ASTR_G];    // A tile, f32 (async copies raw)
  __shared__ bf16_t Bsh[2][32 * BSTR_G];    // B tile, [n][k] row-major

  auto issue = [&](int buf, int k0) {
#if ASYNC_LDS
    #pragma unroll
    for (int c = tid; c < 512; c += 128) {           // A: 512 x 16B chunks
      int r = c >> 3, sgm = c & 7;
      cp_async_b128(&A[(size_t)(m0 + r) * lda + k0 + sgm * 4],
                    &Asf[buf][r * ASTR_G + sgm * 4]);
    }
    {                                                // B: 128 x 16B chunks
      int nn = tid >> 2, sgm = tid & 3;
      cp_async_b128(&W[(size_t)(n0 + nn) * K + k0 + sgm * 8],
                    &Bsh[buf][nn * BSTR_G + sgm * 8]);
    }
#else
    #pragma unroll
    for (int c = tid; c < 64 * 32; c += 128) {
      int r = c >> 5, kk = c & 31;
      Asf[buf][r * ASTR_G + kk] = A[(size_t)(m0 + r) * lda + k0 + kk];
    }
    #pragma unroll
    for (int c = tid; c < 32 * 32; c += 128) {
      int nn = c >> 5, kk = c & 31;
      Bsh[buf][nn * BSTR_G + kk] = W[(size_t)(n0 + nn) * K + k0 + kk];
    }
    if (k0 + 32 < K)
      __builtin_prefetch(&W[(size_t)(n0 + (tid & 31)) * K + k0 + 32], 0, 1);
#endif
  };

  v8f acc0 = {}, acc1 = {};
  const int nsteps = K >> 5;
  issue(0, 0);
  for (int kt = 0; kt < nsteps; ++kt) {
    const int buf = kt & 1;
    if (kt + 1 < nsteps) {
      issue(buf ^ 1, (kt + 1) << 5);                 // keep next tile in flight
      cp_wait<5>();                                  // retire current tile
    } else {
      cp_wait<0>();
    }
    __syncthreads();

    const int grp = lane >> 4;
    const int ml  = wave * 16 + (lane & 15);
    const int nl  = lane & 15;
    v16bf af, b0, b1;
    #pragma unroll
    for (int j = 0; j < 16; ++j) af[j] = (bf16_t)Asf[buf][ml * ASTR_G + a_k_of(j, grp)];
    #pragma unroll
    for (int j = 0; j < 16; ++j) b0[j] = Bsh[buf][nl * BSTR_G + grp * 16 + j];
    #pragma unroll
    for (int j = 0; j < 16; ++j) b1[j] = Bsh[buf][(16 + nl) * BSTR_G + grp * 16 + j];
    acc0 = __builtin_amdgcn_wmma_f32_16x16x32_bf16(false, af, false, b0,
                                                   (short)0, acc0, false, false);
    acc1 = __builtin_amdgcn_wmma_f32_16x16x32_bf16(false, af, false, b1,
                                                   (short)0, acc1, false, false);
    __syncthreads();
  }

  const int nl = lane & 15;
  const int mb = m0 + wave * 16 + (lane >> 4) * 8;
  const float bv0 = bias[n0 + nl];
  const float bv1 = bias[n0 + 16 + nl];
  #pragma unroll
  for (int v = 0; v < 8; ++v) {
    float r0 = acc0[v] + bv0;
    float r1 = acc1[v] + bv1;
    if (RELU) { r0 = fmaxf(r0, 0.0f); r1 = fmaxf(r1, 0.0f); }
    C[(size_t)(mb + v) * Nout + n0 + nl]      = r0;
    C[(size_t)(mb + v) * Nout + n0 + 16 + nl] = r1;
  }
}

// ------------------------------ attention: one (b,h) per block -------------
#define QSTR 34
#define SSTR 130

__global__ void __launch_bounds__(128) k_attention(
    const float* __restrict__ Qg, const float* __restrict__ Kg,
    const float* __restrict__ Vg, float* __restrict__ Og)
{
  const int bh   = blockIdx.x;
  const int b    = bh >> 2;                 // / H_
  const int h    = bh & 3;
  const int tid  = threadIdx.x;
  const int wave = tid >> 5;
  const int lane = tid & 31;

  __shared__ bf16_t Qs[L_ * QSTR];          // 128x32 bf16 (padded)
  __shared__ bf16_t Ks[L_ * QSTR];
  __shared__ bf16_t Vs[L_ * QSTR];
  __shared__ bf16_t Ss[L_ * SSTR];          // scores -> probabilities, bf16

  const size_t base = ((size_t)b * L_) * D_ + (size_t)h * DK_;
  for (int i = tid; i < L_ * DK_; i += 128) {
    int l = i >> 5, c = i & 31;
    size_t g = base + (size_t)l * D_ + c;
    Qs[l * QSTR + c] = (bf16_t)Qg[g];
    Ks[l * QSTR + c] = (bf16_t)Kg[g];
    Vs[l * QSTR + c] = (bf16_t)Vg[g];
  }
  __syncthreads();

  const int grp = lane >> 4;
  const int nl  = lane & 15;
  const float sc = 0.17677669529f;          // 1/sqrt(dk)

  // S = Q K^T : 8x8 = 64 tiles of 16x16, dk=32 -> exactly one WMMA per tile
  for (int t = wave; t < 64; t += 4) {
    int tq = t >> 3, tk = t & 7;
    v16bf af, bmat;
    #pragma unroll
    for (int j = 0; j < 16; ++j) af[j]   = Qs[(tq * 16 + nl) * QSTR + a_k_of(j, grp)];
    #pragma unroll
    for (int j = 0; j < 16; ++j) bmat[j] = Ks[(tk * 16 + nl) * QSTR + (grp * 16 + j)];
    v8f s = {};
    s = __builtin_amdgcn_wmma_f32_16x16x32_bf16(false, af, false, bmat,
                                                (short)0, s, false, false);
    int row0 = tq * 16 + grp * 8;
    int col  = tk * 16 + nl;
    #pragma unroll
    for (int v = 0; v < 8; ++v) Ss[(row0 + v) * SSTR + col] = (bf16_t)(s[v] * sc);
  }
  __syncthreads();

  // softmax: one of the 128 rows per thread, straight from LDS
  {
    int row = tid;
    float mx = -3.0e38f;
    for (int j = 0; j < L_; ++j) mx = fmaxf(mx, (float)Ss[row * SSTR + j]);
    float sum = 0.0f;
    for (int j = 0; j < L_; ++j) sum += __expf((float)Ss[row * SSTR + j] - mx);
    float inv = 1.0f / sum;
    for (int j = 0; j < L_; ++j)
      Ss[row * SSTR + j] = (bf16_t)(__expf((float)Ss[row * SSTR + j] - mx) * inv);
  }
  __syncthreads();

  // O = P V : 8x2 = 16 tiles, K loop over 128 keys in steps of 32
  for (int t = wave; t < 16; t += 4) {
    int tm = t >> 1, tn = t & 1;
    v8f acc = {};
    for (int k0 = 0; k0 < L_; k0 += 32) {
      v16bf af, bmat;
      #pragma unroll
      for (int j = 0; j < 16; ++j) af[j]   = Ss[(tm * 16 + nl) * SSTR + k0 + a_k_of(j, grp)];
      #pragma unroll
      for (int j = 0; j < 16; ++j) bmat[j] = Vs[(k0 + grp * 16 + j) * QSTR + tn * 16 + nl];
      acc = __builtin_amdgcn_wmma_f32_16x16x32_bf16(false, af, false, bmat,
                                                    (short)0, acc, false, false);
    }
    int row0 = tm * 16 + grp * 8;
    int col  = tn * 16 + nl;
    #pragma unroll
    for (int v = 0; v < 8; ++v)
      Og[base + (size_t)(row0 + v) * D_ + col] = acc[v];
  }
}

// ------------------------------ residual + LayerNorm (wave per row) --------
__global__ void __launch_bounds__(256) k_add_ln(
    const float* __restrict__ x, const float* __restrict__ a,
    const float* __restrict__ g, const float* __restrict__ bb,
    float* __restrict__ out)
{
  int row  = blockIdx.x * 8 + (threadIdx.x >> 5);
  int lane = threadIdx.x & 31;
  const size_t rb = (size_t)row * D_;
  float v[4]; float s = 0.0f;
  #pragma unroll
  for (int i = 0; i < 4; ++i) { int d = lane + i * 32; v[i] = x[rb + d] + a[rb + d]; s += v[i]; }
  #pragma unroll
  for (int o = 16; o >= 1; o >>= 1) s += __shfl_xor(s, o, 32);
  float mean = s * (1.0f / D_);
  float var = 0.0f;
  #pragma unroll
  for (int i = 0; i < 4; ++i) { float d0 = v[i] - mean; var += d0 * d0; }
  #pragma unroll
  for (int o = 16; o >= 1; o >>= 1) var += __shfl_xor(var, o, 32);
  var *= (1.0f / D_);
  float inv = rsqrtf(var + 1e-5f);
  #pragma unroll
  for (int i = 0; i < 4; ++i) {
    int d = lane + i * 32;
    out[rb + d] = (v[i] - mean) * inv * g[d] + bb[d];
  }
}

// ------------------------------ mixture head (event_pred) ------------------
__global__ void k_mixture(const float* __restrict__ AL, const float* __restrict__ ML,
                          const float* __restrict__ SL, float* __restrict__ out,
                          unsigned seed)
{
  int i = blockIdx.x * 256 + threadIdx.x;          // over B_*T_
  if (i >= B_ * T_) return;
  int b = i / T_, t = i - b * T_;
  size_t base = (size_t)b * (5 * T_) + t;
  float lg[5]; float mx = -3.0e38f;
  #pragma unroll
  for (int k = 0; k < 5; ++k) { lg[k] = AL[base + (size_t)k * T_]; mx = fmaxf(mx, lg[k]); }
  float se = 0.0f;
  #pragma unroll
  for (int k = 0; k < 5; ++k) { lg[k] = __expf(lg[k] - mx); se += lg[k]; }
  float inv = 1.0f / se, acc = 0.0f;
  #pragma unroll
  for (int k = 0; k < 5; ++k) {
    float mean = ML[base + (size_t)k * T_];
    float sd   = __expf(SL[base + (size_t)k * T_]);
    float nz   = rnorm_hash(seed, (unsigned)(b * 5 + k) * T_ + t);
    acc += lg[k] * inv * (mean + sd * nz);
  }
  out[i] = acc;
}

// ------------------------------ time head (wave per batch) -----------------
__global__ void __launch_bounds__(32) k_time_head(
    const float* __restrict__ X, const float* __restrict__ atw,
    const float* __restrict__ atb,
    const float* __restrict__ tmw, const float* __restrict__ tmb,
    const float* __restrict__ tsw, const float* __restrict__ tsb,
    float* __restrict__ out, unsigned seed)
{
  int b = blockIdx.x, lane = threadIdx.x;
  const float* h = X + (size_t)b * L_ * D_;        // token 0 of batch b
  float lg[5], tm[5], ts[5];
  #pragma unroll
  for (int k = 0; k < 5; ++k) {
    float pa = 0, pm = 0, ps = 0;
    for (int d = lane; d < D_; d += 32) {
      float hv = h[d];
      pa += hv * atw[k * D_ + d];
      pm += hv * tmw[k * D_ + d];
      ps += hv * tsw[k * D_ + d];
    }
    #pragma unroll
    for (int o = 16; o >= 1; o >>= 1) {
      pa += __shfl_xor(pa, o, 32);
      pm += __shfl_xor(pm, o, 32);
      ps += __shfl_xor(ps, o, 32);
    }
    lg[k] = pa + atb[k];
    tm[k] = pm + tmb[k];
    ts[k] = __expf(ps + tsb[k]);
  }
  if (lane == 0) {
    float mx = -3.0e38f;
    for (int k = 0; k < 5; ++k) mx = fmaxf(mx, lg[k]);
    float se = 0.0f;
    for (int k = 0; k < 5; ++k) { lg[k] = __expf(lg[k] - mx); se += lg[k]; }
    float acc = 0.0f;
    for (int k = 0; k < 5; ++k)
      acc += (lg[k] / se) * (tm[k] + ts[k] * rnorm_hash(seed, (unsigned)(b * 5 + k)));
    out[b] = acc;
  }
}

// ---------------------------------------------------------------------------
extern "C" void kernel_launch(void* const* d_in, const int* in_sizes, int n_in,
                              void* d_out, int out_size, void* d_ws, size_t ws_size,
                              hipStream_t stream)
{
  (void)in_sizes; (void)n_in; (void)out_size; (void)ws_size;

  const int*   ev     = (const int*)  d_in[0];
  const int*   maskp  = (const int*)  d_in[2];
  const float* emb    = (const float*)d_in[4];
  const float* dwmu   = (const float*)d_in[5];
  const float* dwls   = (const float*)d_in[6];
  const float* dbmu   = (const float*)d_in[7];
  const float* dbls   = (const float*)d_in[8];
  const float* f1wmu  = (const float*)d_in[9];
  const float* f1wls  = (const float*)d_in[10];
  const float* f1bmu  = (const float*)d_in[11];
  const float* f1bls  = (const float*)d_in[12];
  const float* f2wmu  = (const float*)d_in[13];
  const float* f2wls  = (const float*)d_in[14];
  const float* f2bmu  = (const float*)d_in[15];
  const float* f2bls  = (const float*)d_in[16];
  const float* ln_g   = (const float*)d_in[17];
  const float* ln_b   = (const float*)d_in[18];
  const float* alwmu  = (const float*)d_in[19];
  const float* alwls  = (const float*)d_in[20];
  const float* albmu  = (const float*)d_in[21];
  const float* albls  = (const float*)d_in[22];
  const float* atwmu  = (const float*)d_in[23];
  const float* atwls  = (const float*)d_in[24];
  const float* atbmu  = (const float*)d_in[25];
  const float* atbls  = (const float*)d_in[26];
  const float* em_w   = (const float*)d_in[27];
  const float* em_b   = (const float*)d_in[28];
  const float* es_w   = (const float*)d_in[29];
  const float* es_b   = (const float*)d_in[30];
  const float* tm_w   = (const float*)d_in[31];
  const float* tm_b   = (const float*)d_in[32];
  const float* ts_w   = (const float*)d_in[33];
  const float* ts_b   = (const float*)d_in[34];

  // ---- carve workspace (~368 MB) ----
  char* wsp = (char*)d_ws;
  size_t off = 0;
  auto alloc = [&](size_t bytes) -> void* {
    void* p = wsp + off;
    off = (off + bytes + 255) & ~(size_t)255;
    return p;
  };
  const size_t ND = (size_t)NT_ * D_;
  float*  X    = (float*) alloc(ND * 4);
  float*  INP  = (float*) alloc(ND * 4);
  float*  Qb   = (float*) alloc(ND * 4);
  float*  Kb   = (float*) alloc(ND * 4);
  float*  Vb   = (float*) alloc(ND * 4);
  float*  ATT  = (float*) alloc(ND * 4);
  float*  AO   = (float*) alloc(ND * 4);
  float*  H1   = (float*) alloc((size_t)NT_ * DFF_ * 4);
  float*  AL   = (float*) alloc((size_t)B_ * 5 * T_ * 4);
  float*  ML   = (float*) alloc((size_t)B_ * 5 * T_ * 4);
  float*  SL   = (float*) alloc((size_t)B_ * 5 * T_ * 4);
  bf16_t* Wbuf = (bf16_t*)alloc((size_t)5 * T_ * D_ * 2);   // largest sampled W
  float*  Bbuf = (float*) alloc((size_t)5 * T_ * 4);
  float*  ATW  = (float*) alloc((size_t)5 * D_ * 4);
  float*  ATB  = (float*) alloc(5 * 4);

  unsigned seedc = 0x1234567u;
  auto nseed = [&]() { return seedc++; };

  auto sampled_gemm = [&](const float* Amat, int lda,
                          const float* wmu, const float* wls,
                          const float* bmu, const float* bls,
                          int M, int K, int Nout, float* C, bool relu) {
    int wn = Nout * K;
    k_sample_bf16<<<dim3((wn + 255) / 256), dim3(256), 0, stream>>>(wmu, wls, Wbuf, wn, nseed());
    k_sample_f32 <<<dim3((Nout + 255) / 256), dim3(256), 0, stream>>>(bmu, bls, Bbuf, Nout, nseed());
    dim3 g(Nout / 32, M / 64);
    if (relu) k_gemm_bf16<true ><<<g, 128, 0, stream>>>(Amat, lda, Wbuf, Bbuf, C, K, Nout);
    else      k_gemm_bf16<false><<<g, 128, 0, stream>>>(Amat, lda, Wbuf, Bbuf, C, K, Nout);
  };

  // ---- embedding + positional encoding ----
  k_embed_pos<<<dim3((NT_ * D_ + 255) / 256), dim3(256), 0, stream>>>(ev, maskp, emb, X, INP);

  const size_t DD = (size_t)D_ * D_;
  for (int l = 0; l < 2; ++l) {
    const float* wmu = dwmu + (size_t)l * 8 * DD;
    const float* wls = dwls + (size_t)l * 8 * DD;
    const float* bmu = dbmu + (size_t)l * 8 * D_;
    const float* bls = dbls + (size_t)l * 8 * D_;

    // self-attention on x
    sampled_gemm(X,  D_, wmu + 0 * DD, wls + 0 * DD, bmu + 0 * D_, bls + 0 * D_, NT_, D_, D_, Qb, false);
    sampled_gemm(X,  D_, wmu + 1 * DD, wls + 1 * DD, bmu + 1 * D_, bls + 1 * D_, NT_, D_, D_, Kb, false);
    sampled_gemm(X,  D_, wmu + 2 * DD, wls + 2 * DD, bmu + 2 * D_, bls + 2 * D_, NT_, D_, D_, Vb, false);
    k_attention<<<dim3(B_ * H_), dim3(128), 0, stream>>>(Qb, Kb, Vb, ATT);
    sampled_gemm(ATT, D_, wmu + 3 * DD, wls + 3 * DD, bmu + 3 * D_, bls + 3 * D_, NT_, D_, D_, AO, false);
    k_add_ln<<<dim3(NT_ / 8), dim3(256), 0, stream>>>(X, AO, ln_g + (l * 3 + 0) * D_, ln_b + (l * 3 + 0) * D_, X);

    // cross-attention: Q from x, K/V from inp_x
    sampled_gemm(X,   D_, wmu + 4 * DD, wls + 4 * DD, bmu + 4 * D_, bls + 4 * D_, NT_, D_, D_, Qb, false);
    sampled_gemm(INP, D_, wmu + 5 * DD, wls + 5 * DD, bmu + 5 * D_, bls + 5 * D_, NT_, D_, D_, Kb, false);
    sampled_gemm(INP, D_, wmu + 6 * DD, wls + 6 * DD, bmu + 6 * D_, bls + 6 * D_, NT_, D_, D_, Vb, false);
    k_attention<<<dim3(B_ * H_), dim3(128), 0, stream>>>(Qb, Kb, Vb, ATT);
    sampled_gemm(ATT, D_, wmu + 7 * DD, wls + 7 * DD, bmu + 7 * D_, bls + 7 * D_, NT_, D_, D_, AO, false);
    k_add_ln<<<dim3(NT_ / 8), dim3(256), 0, stream>>>(X, AO, ln_g + (l * 3 + 1) * D_, ln_b + (l * 3 + 1) * D_, X);

    // FFN
    sampled_gemm(X,  D_,   f1wmu + (size_t)l * DFF_ * D_, f1wls + (size_t)l * DFF_ * D_,
                 f1bmu + (size_t)l * DFF_, f1bls + (size_t)l * DFF_,
                 NT_, D_, DFF_, H1, true);
    sampled_gemm(H1, DFF_, f2wmu + (size_t)l * D_ * DFF_, f2wls + (size_t)l * D_ * DFF_,
                 f2bmu + (size_t)l * D_, f2bls + (size_t)l * D_,
                 NT_, DFF_, D_, AO, false);
    k_add_ln<<<dim3(NT_ / 8), dim3(256), 0, stream>>>(X, AO, ln_g + (l * 3 + 2) * D_, ln_b + (l * 3 + 2) * D_, X);
  }

  // ---- heads: h = X[:,0,:] read in-place via lda = L_*D_ ----
  const int NH = 5 * T_;
  sampled_gemm(X, L_ * D_, alwmu, alwls, albmu, albls, B_, D_, NH, AL, false);

  dim3 gh(NH / 32, B_ / 64);
  k_cvt_bf16<<<dim3((NH * D_ + 255) / 256), dim3(256), 0, stream>>>(em_w, Wbuf, NH * D_);
  k_gemm_bf16<false><<<gh, 128, 0, stream>>>(X, L_ * D_, Wbuf, em_b, ML, D_, NH);
  k_cvt_bf16<<<dim3((NH * D_ + 255) / 256), dim3(256), 0, stream>>>(es_w, Wbuf, NH * D_);
  k_gemm_bf16<false><<<gh, 128, 0, stream>>>(X, L_ * D_, Wbuf, es_b, SL, D_, NH);

  k_mixture<<<dim3((B_ * T_ + 255) / 256), dim3(256), 0, stream>>>(AL, ML, SL, (float*)d_out, nseed());

  // ---- time head ----
  k_sample_f32<<<dim3((5 * D_ + 255) / 256), dim3(256), 0, stream>>>(atwmu, atwls, ATW, 5 * D_, nseed());
  k_sample_f32<<<dim3(1), dim3(256), 0, stream>>>(atbmu, atbls, ATB, 5, nseed());
  k_time_head<<<dim3(B_), dim3(32), 0, stream>>>(X, ATW, ATB, tm_w, tm_b, ts_w, ts_b,
                                                 (float*)d_out + (size_t)B_ * T_, nseed());
}
